// AuxFormerXYZ_31568009625960
// MI455X (gfx1250) — matrix operand order repaired and checked by reading, not compiled, not run on hardware
//
#include <hip/hip_runtime.h>

// ---------------- structural constants ----------------
constexpr int BATCH = 32;
constexpr int TT    = 75;     // timesteps
constexpr int JJ    = 32;     // joints
constexpr int HH    = 8;      // heads
constexpr int NLAYER = 16;
constexpr int NTOK  = BATCH * JJ * TT;       // 76800
constexpr float ATTN_SCALE = 0.125f;         // 64^-0.5
constexpr float LN_EPS = 1e-5f;

typedef __attribute__((ext_vector_type(16))) __bf16 v16bf;
typedef __attribute__((ext_vector_type(8)))  float  v8f;

union FragU { v16bf v; unsigned u[8]; unsigned short s[16]; uint4 q[2]; };

__device__ inline unsigned short f2bf(float x) {
  union { float f; unsigned u; } a; a.f = x;
  unsigned r = a.u + 0x7FFFu + ((a.u >> 16) & 1u);
  return (unsigned short)(r >> 16);
}

// A fragment: 16x32 bf16, row-major source, leading dim ld.
// lane<16 -> row=lane, K in {0..7,16..23}; lane>=16 -> row=lane-16, K in {8..15,24..31}.
__device__ inline v16bf ld_afrag(const unsigned short* base, int row0, int ld, int k0) {
  const int lane = threadIdx.x & 31;
  const unsigned short* rp = base + (size_t)(row0 + (lane & 15)) * ld + k0 + ((lane >> 4) << 3);
  FragU f;
  f.q[0] = *(const uint4*)rp;
  f.q[1] = *(const uint4*)(rp + 16);
  return f.v;
}

// B fragment: source stored transposed (N x K row-major, leading dim ld).
// lane<16 -> col=lane, K in {0..7,16..23}; lane>=16 -> col=lane-16, K in {8..15,24..31}.
__device__ inline v16bf ld_bfrag(const unsigned short* base, int col0, int ld, int k0) {
  const int lane = threadIdx.x & 31;
  const unsigned short* rp = base + (size_t)(col0 + (lane & 15)) * ld + k0 + ((lane >> 4) << 3);
  FragU f;
  f.q[0] = *(const uint4*)rp;
  f.q[1] = *(const uint4*)(rp + 16);
  return f.v;
}

__device__ inline v8f wmma_bf16(v16bf a, v16bf b, v8f c) {
  return __builtin_amdgcn_wmma_f32_16x16x32_bf16(false, a, false, b, (short)0, c, false, false);
}

// ---------------- async global -> LDS (CDNA5) ----------------
__device__ inline void async_ld16(const unsigned short* lds, const unsigned short* g) {
  unsigned loff = (unsigned)(unsigned long long)(const void*)lds;  // flat->LDS uses addr[31:0]
  unsigned long long ga = (unsigned long long)(const void*)g;
  asm volatile("global_load_async_to_lds_b128 %0, %1, off" :: "v"(loff), "v"(ga) : "memory");
}
__device__ inline void wait_async6() { asm volatile("s_wait_asynccnt 6" ::: "memory"); }
__device__ inline void wait_async0() { asm volatile("s_wait_asynccnt 0" ::: "memory"); }

// ---------------- weight convert + transpose: (L,K,N) f32 -> (L,N,K) bf16 ----------------
__global__ void convtrans_kernel(const float* __restrict__ src, unsigned short* __restrict__ dst,
                                 int K, int N, int total) {
  int tid = blockIdx.x * blockDim.x + threadIdx.x;
  if (tid >= total) return;
  int per = K * N;
  int l = tid / per, rem = tid - l * per;
  int k = rem / N, n = rem - k * N;
  dst[(size_t)l * per + (size_t)n * K + k] = f2bf(src[tid]);
}

// ---------------- patch embed + pos + agent ----------------
__global__ void embed_kernel(const float* __restrict__ x, const float* __restrict__ Wp,
                             const float* __restrict__ bp, const float* __restrict__ pos,
                             const float* __restrict__ agent, float* __restrict__ h) {
  int tid = blockIdx.x * blockDim.x + threadIdx.x;   // NTOK*128 threads
  int d = tid & 127, tok = tid >> 7;
  int b = tok / (JJ * TT); int rem = tok - b * (JJ * TT);
  int n = rem / TT; int t = rem - n * TT;
  const float* xp = x + ((size_t)(b * TT + t) * JJ + n) * 3;
  h[tid] = xp[0] * Wp[d] + xp[1] * Wp[128 + d] + xp[2] * Wp[256 + d]
         + bp[d] + pos[t * 128 + d] + agent[n * 128 + d];
}

__global__ void add_dec_kernel(float* __restrict__ h, const float* __restrict__ dec_pos,
                               const float* __restrict__ dec_agent) {
  int tid = blockIdx.x * blockDim.x + threadIdx.x;
  int d = tid & 127, tok = tid >> 7;
  int rem = tok % (JJ * TT);
  int n = rem / TT; int t = rem - n * TT;
  h[tid] += dec_pos[t * 128 + d] + dec_agent[n * 128 + d];
}

// ---------------- LayerNorm: one wave per token, fp32 in -> bf16 out ----------------
__global__ void ln_kernel(const float* __restrict__ h, const float* __restrict__ g,
                          const float* __restrict__ b, unsigned short* __restrict__ z) {
  int tok  = blockIdx.x * 8 + (threadIdx.x >> 5);
  int lane = threadIdx.x & 31;
  float4 v = ((const float4*)(h + (size_t)tok * 128))[lane];
  float sum = v.x + v.y + v.z + v.w;
#pragma unroll
  for (int off = 16; off > 0; off >>= 1) sum += __shfl_xor(sum, off, 32);
  float mu = sum * (1.0f / 128.0f);
  float dx = v.x - mu, dy = v.y - mu, dz = v.z - mu, dw = v.w - mu;
  float var = dx * dx + dy * dy + dz * dz + dw * dw;
#pragma unroll
  for (int off = 16; off > 0; off >>= 1) var += __shfl_xor(var, off, 32);
  float rs = rsqrtf(var * (1.0f / 128.0f) + LN_EPS);
  float4 g4 = ((const float4*)g)[lane];
  float4 b4 = ((const float4*)b)[lane];
  union { uint2 p; unsigned short s[4]; } o;
  o.s[0] = f2bf(dx * rs * g4.x + b4.x);
  o.s[1] = f2bf(dy * rs * g4.y + b4.y);
  o.s[2] = f2bf(dz * rs * g4.z + b4.z);
  o.s[3] = f2bf(dw * rs * g4.w + b4.w);
  ((uint2*)(z + (size_t)tok * 128))[lane] = o.p;
}

// ---------------- WMMA GEMM with async-LDS double buffering ----------------
// Block: 256 threads (8 waves). Block tile 256(M) x 128(N); wave tile 32 x 128.
// C(MxN) = A(MxK,bf16) * Bt(NxK,bf16). K multiple of 32, N multiple of 128, M multiple of 256.
// MODE 0: store bf16. MODE 1: +bias, exact GELU, bf16. MODE 2: +bias, += into fp32 out.
template<int MODE>
__global__ __launch_bounds__(256) void gemm_bf16(
    const unsigned short* __restrict__ A, const unsigned short* __restrict__ Bt,
    const float* __restrict__ bias, void* __restrict__ out, int N, int K) {
  __shared__ unsigned short As[2][256 * 32];
  __shared__ unsigned short Bs[2][128 * 32];
  const int tid = threadIdx.x;
  const int wave = tid >> 5;
  const int rowBlk = blockIdx.y * 256;
  const int colBlk = blockIdx.x * 128;
  const int wrow = wave * 32;

  v8f zero = {};
  v8f c[2][8];
#pragma unroll
  for (int mi = 0; mi < 2; ++mi)
#pragma unroll
    for (int ni = 0; ni < 8; ++ni) c[mi][ni] = zero;

  // stage issue: A tile 256x32 (1024 16B chunks), B tile 128x32 (512 chunks); 6 per thread
  auto stage = [&](int k0, unsigned short* Ad, unsigned short* Bd) {
#pragma unroll
    for (int i = 0; i < 4; ++i) {
      int ch = tid + i * 256, r = ch >> 2, cc = ch & 3;
      async_ld16(Ad + r * 32 + cc * 8, A + (size_t)(rowBlk + r) * K + k0 + cc * 8);
    }
#pragma unroll
    for (int i = 0; i < 2; ++i) {
      int ch = tid + i * 256, r = ch >> 2, cc = ch & 3;
      async_ld16(Bd + r * 32 + cc * 8, Bt + (size_t)(colBlk + r) * K + k0 + cc * 8);
    }
  };

  const int nsteps = K >> 5;
  stage(0, As[0], Bs[0]);
  for (int s = 0; s < nsteps; ++s) {
    if (s + 1 < nsteps) { stage((s + 1) << 5, As[(s + 1) & 1], Bs[(s + 1) & 1]); wait_async6(); }
    else wait_async0();
    __syncthreads();
    const unsigned short* Ab = As[s & 1];
    const unsigned short* Bb = Bs[s & 1];
    v16bf a0 = ld_afrag(Ab, wrow, 32, 0);
    v16bf a1 = ld_afrag(Ab, wrow + 16, 32, 0);
#pragma unroll
    for (int ni = 0; ni < 8; ++ni) {
      v16bf b = ld_bfrag(Bb, ni * 16, 32, 0);
      c[0][ni] = wmma_bf16(a0, b, c[0][ni]);
      c[1][ni] = wmma_bf16(a1, b, c[1][ni]);
    }
    __syncthreads();
  }

  const int lane = tid & 31;
  const int nn = lane & 15, mh = lane >> 4;
#pragma unroll
  for (int mi = 0; mi < 2; ++mi)
#pragma unroll
    for (int ni = 0; ni < 8; ++ni) {
      int col = colBlk + ni * 16 + nn;
#pragma unroll
      for (int r = 0; r < 8; ++r) {
        int row = rowBlk + wrow + mi * 16 + mh * 8 + r;
        size_t idx = (size_t)row * N + col;
        float v = c[mi][ni][r];
        if (MODE == 0) {
          ((unsigned short*)out)[idx] = f2bf(v);
        } else if (MODE == 1) {
          v += bias[col];
          v = 0.5f * v * (1.0f + erff(v * 0.70710678118654752f));
          ((unsigned short*)out)[idx] = f2bf(v);
        } else {
          ((float*)out)[idx] += v + bias[col];
        }
      }
    }
}

// ---------------- attention: one workgroup (4 waves) per (sequence, head) ----------------
// LS is compile-time sequence length (75 temporal / 32 spatial).
// MODE 0 (temporal): token = s*LS + e, stride 1. MODE 1 (spatial): token = b*J*T + e*T + t.
template<int LS, int MODE>
__global__ __launch_bounds__(128) void attn_kernel(
    const unsigned short* __restrict__ qkv, unsigned short* __restrict__ o,
    const float* __restrict__ mask, int Tdim) {
  constexpr int LP  = (LS + 15) & ~15;
  constexpr int LPK = (LP & 31) ? (LP + 16) : LP;
  constexpr int NT  = LP >> 4;
  extern __shared__ char smem[];
  unsigned short* Qs = (unsigned short*)smem;          // LP x 64, row-major
  unsigned short* Ks = Qs + LP * 64;                   // LP x 64, row-major
  unsigned short* Vt = Ks + LP * 64;                   // 64 x LPK, V transposed
  float*          S  = (float*)(Vt + 64 * LPK);        // LP x LP
  unsigned short* P  = (unsigned short*)(S + LP * LP); // LP x LPK

  const int s = blockIdx.x, hh = blockIdx.y;
  size_t base; int stride;
  if (MODE == 0) { base = (size_t)s * LS; stride = 1; }
  else { int b = s / Tdim, t = s - b * Tdim; base = (size_t)b * LS * Tdim + t; stride = Tdim; }

  const int tid = threadIdx.x;
  const uint4 z4 = {0u, 0u, 0u, 0u};

  // Q/K rows < LS: async global->LDS (8 x 16B chunks per row)
  for (int i = tid; i < LS * 8; i += 128) {
    int e = i >> 3, u = i & 7;
    async_ld16(Qs + i * 8, qkv + (base + (size_t)e * stride) * 1536 + hh * 64 + u * 8);
  }
  for (int i = tid; i < LS * 8; i += 128) {
    int e = i >> 3, u = i & 7;
    async_ld16(Ks + i * 8, qkv + (base + (size_t)e * stride) * 1536 + 512 + hh * 64 + u * 8);
  }
  // padding rows of Q/K -> zeros
  for (int i = LS * 8 + tid; i < LP * 8; i += 128) {
    *(uint4*)(Qs + (size_t)i * 8) = z4;
    *(uint4*)(Ks + (size_t)i * 8) = z4;
  }
  // V: sync load, transpose into Vt[n][k] (covers all k < LPK; k >= LS -> zeros)
  for (int i = tid; i < LPK * 8; i += 128) {
    int e = i >> 3, u = i & 7; uint4 val = z4;
    if (e < LS) val = *(const uint4*)(qkv + (base + (size_t)e * stride) * 1536 + 1024 + hh * 64 + u * 8);
    const unsigned short* sv = (const unsigned short*)&val;
#pragma unroll
    for (int j = 0; j < 8; ++j) Vt[(u * 8 + j) * LPK + e] = sv[j];
  }
  // zero P
  for (int i = tid; i < (LP * LPK) >> 3; i += 128) *(uint4*)(P + (size_t)i * 8) = z4;
  wait_async0();
  __syncthreads();

  const int wave = tid >> 5;
  // S = Q * K^T (raw logits; scale applied in softmax)
  for (int tile = wave; tile < NT * NT; tile += 4) {
    int ti = tile / NT, tj = tile - ti * NT;
    v8f c = {};
#pragma unroll
    for (int kk = 0; kk < 64; kk += 32)
      c = wmma_bf16(ld_afrag(Qs, ti * 16, 64, kk), ld_bfrag(Ks, tj * 16, 64, kk), c);
    int lane = tid & 31, nn = lane & 15, mh = lane >> 4;
#pragma unroll
    for (int r = 0; r < 8; ++r)
      S[(ti * 16 + mh * 8 + r) * LP + tj * 16 + nn] = c[r];
  }
  __syncthreads();

  // masked, renormalized softmax: p = e*m / (sum(e*m) + 1e-10*sum(e))
  for (int r = tid; r < LS; r += 128) {
    float* Sr = S + r * LP;
    float mx = -3.0e38f;
    for (int ci = 0; ci < LS; ++ci) mx = fmaxf(mx, Sr[ci] * ATTN_SCALE);
    float sum = 0.f;
    for (int ci = 0; ci < LS; ++ci) { float e = __expf(Sr[ci] * ATTN_SCALE - mx); Sr[ci] = e; sum += e; }
    float denom;
    if (mask) {
      const float* mr = mask + ((size_t)s * LS + r) * LS;
      float ms = 0.f;
      for (int ci = 0; ci < LS; ++ci) { float tm = Sr[ci] * mr[ci]; Sr[ci] = tm; ms += tm; }
      denom = ms + 1e-10f * sum;
    } else denom = sum;
    float inv = 1.0f / denom;
    for (int ci = 0; ci < LS; ++ci) P[r * LPK + ci] = f2bf(Sr[ci] * inv);
  }
  __syncthreads();

  // O = P * V   (M=LP, N=64, K=LPK; padded P rows/cols and Vt cols are zero)
  for (int tile = wave; tile < NT * 4; tile += 4) {
    int ti = tile >> 2, tj = tile & 3;
    v8f c = {};
#pragma unroll
    for (int kk = 0; kk < LPK; kk += 32)
      c = wmma_bf16(ld_afrag(P, ti * 16, LPK, kk), ld_bfrag(Vt, tj * 16, LPK, kk), c);
    int lane = tid & 31, nn = lane & 15, mh = lane >> 4;
#pragma unroll
    for (int r = 0; r < 8; ++r) {
      int e = ti * 16 + mh * 8 + r;
      if (e < LS)
        o[(base + (size_t)e * stride) * 512 + hh * 64 + tj * 16 + nn] = f2bf(c[r]);
    }
  }
}

// ---------------- head: out(B,T,J,3) = h @ headW + headb ----------------
__global__ void head_kernel(const float* __restrict__ h, const float* __restrict__ W,
                            const float* __restrict__ bias, float* __restrict__ out) {
  int tid = blockIdx.x * blockDim.x + threadIdx.x;
  if (tid >= NTOK * 3) return;
  int i = tid % 3, tok = tid / 3;
  int b = tok / (JJ * TT); int rem = tok - b * (JJ * TT);
  int n = rem / TT; int t = rem - n * TT;
  const float* hp = h + (size_t)tok * 128;
  float acc = bias[i];
#pragma unroll 4
  for (int d = 0; d < 128; ++d) acc += hp[d] * W[d * 3 + i];
  out[((size_t)(b * TT + t) * JJ + n) * 3 + i] = acc;
}

// ---------------- host orchestration ----------------
extern "C" void kernel_launch(void* const* d_in, const int* in_sizes, int n_in,
                              void* d_out, int out_size, void* d_ws, size_t ws_size,
                              hipStream_t stream) {
  const float* x        = (const float*)d_in[0];
  const float* mask_t   = (const float*)d_in[1];
  const float* Wp       = (const float*)d_in[2];
  const float* bp       = (const float*)d_in[3];
  const float* pos      = (const float*)d_in[4];
  const float* agent    = (const float*)d_in[5];
  const float* dec_pos  = (const float*)d_in[6];
  const float* dec_agent= (const float*)d_in[7];
  const float* headW    = (const float*)d_in[8];
  const float* headb    = (const float*)d_in[9];
  const float* ln1_g    = (const float*)d_in[10];
  const float* ln1_b    = (const float*)d_in[11];
  const float* Wqkv     = (const float*)d_in[12];
  const float* Wout     = (const float*)d_in[13];
  const float* bout     = (const float*)d_in[14];
  const float* ln2_g    = (const float*)d_in[15];
  const float* ln2_b    = (const float*)d_in[16];
  const float* W1       = (const float*)d_in[17];
  const float* bf1      = (const float*)d_in[18];
  const float* W2       = (const float*)d_in[19];
  const float* bf2      = (const float*)d_in[20];

  // workspace carve-out (256B aligned)
  char* wsb = (char*)d_ws;
  size_t off = 0;
  auto carve = [&](size_t bytes) { char* p = wsb + off; off += (bytes + 255) & ~(size_t)255; return p; };
  float*          hbuf   = (float*)         carve((size_t)NTOK * 128 * 4);
  unsigned short* zbuf   = (unsigned short*)carve((size_t)NTOK * 128 * 2);
  unsigned short* qkvbuf = (unsigned short*)carve((size_t)NTOK * 1536 * 2); // reused as FFN act
  unsigned short* obuf   = (unsigned short*)carve((size_t)NTOK * 512 * 2);
  unsigned short* wq_t   = (unsigned short*)carve((size_t)NLAYER * 1536 * 128 * 2);
  unsigned short* wo_t   = (unsigned short*)carve((size_t)NLAYER * 128 * 512 * 2);
  unsigned short* w1_t   = (unsigned short*)carve((size_t)NLAYER * 512 * 128 * 2);
  unsigned short* w2_t   = (unsigned short*)carve((size_t)NLAYER * 128 * 512 * 2);
  unsigned short* actbuf = qkvbuf;
  (void)ws_size; (void)in_sizes; (void)n_in; (void)out_size;

  // weights -> bf16 transposed
  {
    int tq = NLAYER * 128 * 1536;
    convtrans_kernel<<<(tq + 255) / 256, 256, 0, stream>>>(Wqkv, wq_t, 128, 1536, tq);
    int to = NLAYER * 512 * 128;
    convtrans_kernel<<<(to + 255) / 256, 256, 0, stream>>>(Wout, wo_t, 512, 128, to);
    int t1 = NLAYER * 128 * 512;
    convtrans_kernel<<<(t1 + 255) / 256, 256, 0, stream>>>(W1, w1_t, 128, 512, t1);
    int t2 = NLAYER * 512 * 128;
    convtrans_kernel<<<(t2 + 255) / 256, 256, 0, stream>>>(W2, w2_t, 512, 128, t2);
  }

  embed_kernel<<<(NTOK * 128) / 256, 256, 0, stream>>>(x, Wp, bp, pos, agent, hbuf);

  auto smem_for = [](int Ls) {
    int Lp = (Ls + 15) & ~15;
    int Lpk = (Lp & 31) ? (Lp + 16) : Lp;
    return (size_t)(2 * Lp * 64 * 2 + (size_t)64 * Lpk * 2 + (size_t)Lp * Lp * 4 + (size_t)Lp * Lpk * 2);
  };
  const size_t smemT = smem_for(TT);   // temporal: 73728 B
  const size_t smemS = smem_for(JJ);   // spatial: 18432 B
  hipFuncSetAttribute((const void*)attn_kernel<TT, 0>,
                      hipFuncAttributeMaxDynamicSharedMemorySize, (int)smemT);

  const dim3 blk256(256), blk128(128);
  const int MB = NTOK / 256;  // 300 row blocks

  for (int l = 0; l < NLAYER; ++l) {
    const bool temporal = ((l & 1) == 0);
    const bool use_mask = temporal && (l < 8);   // encoder temporal layers only

    // LN1 -> z
    ln_kernel<<<NTOK / 8, blk256, 0, stream>>>(hbuf, ln1_g + l * 128, ln1_b + l * 128, zbuf);
    // qkv = z @ Wqkv[l]   (76800 x 1536, K=128) -> bf16
    gemm_bf16<0><<<dim3(1536 / 128, MB), blk256, 0, stream>>>(
        zbuf, wq_t + (size_t)l * 1536 * 128, nullptr, qkvbuf, 1536, 128);
    // attention
    if (temporal)
      attn_kernel<TT, 0><<<dim3(BATCH * JJ, HH), blk128, smemT, stream>>>(
          qkvbuf, obuf, use_mask ? mask_t : nullptr, TT);
    else
      attn_kernel<JJ, 1><<<dim3(BATCH * TT, HH), blk128, smemS, stream>>>(
          qkvbuf, obuf, nullptr, TT);
    // h += o @ Wout[l] + bout[l]   (K=512, N=128)
    gemm_bf16<2><<<dim3(1, MB), blk256, 0, stream>>>(
        obuf, wo_t + (size_t)l * 128 * 512, bout + l * 128, hbuf, 128, 512);
    // LN2 -> z
    ln_kernel<<<NTOK / 8, blk256, 0, stream>>>(hbuf, ln2_g + l * 128, ln2_b + l * 128, zbuf);
    // act = gelu(z @ W1[l] + bf1[l])   (N=512, K=128) -> bf16
    gemm_bf16<1><<<dim3(512 / 128, MB), blk256, 0, stream>>>(
        zbuf, w1_t + (size_t)l * 512 * 128, bf1 + l * 512, actbuf, 512, 128);
    // h += act @ W2[l] + bf2[l]   (N=128, K=512)
    gemm_bf16<2><<<dim3(1, MB), blk256, 0, stream>>>(
        actbuf, w2_t + (size_t)l * 128 * 512, bf2 + l * 128, hbuf, 128, 512);

    if (l == 7)  // between encoder and decoder stacks
      add_dec_kernel<<<(NTOK * 128) / 256, blk256, 0, stream>>>(hbuf, dec_pos, dec_agent);
  }

  head_kernel<<<(NTOK * 3 + 255) / 256, blk256, 0, stream>>>(hbuf, headW, headb, (float*)d_out);
}
